// SynthesizerAttention_15590731284880
// MI455X (gfx1250) — compile-verified, matrix operand order checked by reading, not run on hardware
//
#include <hip/hip_runtime.h>

typedef __bf16 bf16;
typedef __attribute__((ext_vector_type(16))) __bf16 v16bf;
typedef __attribute__((ext_vector_type(8)))  __bf16 v8bf;
typedef __attribute__((ext_vector_type(4)))  __bf16 v4bf;
typedef __attribute__((ext_vector_type(8)))  float   v8f;
typedef __attribute__((ext_vector_type(4)))  float   v4f;
typedef __attribute__((ext_vector_type(4)))  int     v4i;

#define WMMA_BF16(a, b, c) \
  __builtin_amdgcn_wmma_f32_16x16x32_bf16(false, (a), false, (b), (short)0, (c), false, false)

// CDNA5 async global->LDS DMA (ASYNCcnt-tracked). Guarded so a missing builtin
// falls back to the synchronous load+ds_store path. Probe-learned signature:
// (int4 global*, int4 shared*, imm offset, imm cpol).
#if defined(__HIP_DEVICE_COMPILE__) && \
    __has_builtin(__builtin_amdgcn_global_load_async_to_lds_b128)
#define HAVE_ASYNC_LDS 1
#else
#define HAVE_ASYNC_LDS 0
#endif

static constexpr int Bc = 4, Sc = 1024, Fc = 1024, Hc = 16, Dc = 64;
static constexpr int BSc = Bc * Sc;  // 4096 rows

__device__ __forceinline__ void cp_b128(bf16* lds, const bf16* g) {
#if HAVE_ASYNC_LDS
  __builtin_amdgcn_global_load_async_to_lds_b128(
      (__attribute__((address_space(1))) v4i*)(void*)g,
      (__attribute__((address_space(3))) v4i*)(void*)lds, /*offset=*/0, /*cpol=*/0);
#else
  *(v8bf*)lds = *(const v8bf*)g;
#endif
}

__device__ __forceinline__ void wait_async_copies() {
#if HAVE_ASYNC_LDS
#if __has_builtin(__builtin_amdgcn_s_wait_asynccnt)
  __builtin_amdgcn_s_wait_asynccnt(0);
#else
  asm volatile("s_wait_asynccnt 0x0" ::: "memory");
#endif
#endif
}

// Build a v16bf A/B fragment for v_wmma_f32_16x16x32_bf16 from a row-major
// 16x(>=32) bf16 tile with row stride ldk. Matches the ISA 16-bit layout:
// lanes 0-15: row=lane, K {0..7} in v[0..7], K {16..23} in v[8..15];
// lanes 16-31: row=lane-16, K {8..15}, K {24..31}.
__device__ __forceinline__ v16bf load_frag(const bf16* __restrict__ base, int ldk, int lane) {
  int r = lane & 15;
  int koff = (lane & 16) ? 8 : 0;
  const bf16* p = base + (size_t)r * ldk + koff;
  v8bf lo = *(const v8bf*)(p);
  v8bf hi = *(const v8bf*)(p + 16);
  v16bf f;
#pragma unroll
  for (int i = 0; i < 8; ++i) { f[i] = lo[i]; f[i + 8] = hi[i]; }
  return f;
}

// ---------------- conversion kernels ----------------

__global__ __launch_bounds__(256) void cvt_bf16_kernel(const float* __restrict__ src,
                                                       bf16* __restrict__ dst, int n4) {
  int i = blockIdx.x * 256 + threadIdx.x;
  if (i >= n4) return;
  v4f x = *(const v4f*)(src + (size_t)i * 4);
  v4bf y;
#pragma unroll
  for (int j = 0; j < 4; ++j) y[j] = (bf16)x[j];
  *(v4bf*)(dst + (size_t)i * 4) = y;
}

// Wt[n][k] = (bf16)W[k][n] for four 1024x1024 matrices (z selects which).
__global__ __launch_bounds__(256) void transpose_cvt_kernel(
    const float* __restrict__ W0, const float* __restrict__ W1,
    const float* __restrict__ W2, const float* __restrict__ W3,
    bf16* __restrict__ T0, bf16* __restrict__ T1,
    bf16* __restrict__ T2, bf16* __restrict__ T3) {
  int z = blockIdx.z;
  const float* W = (z == 0) ? W0 : (z == 1) ? W1 : (z == 2) ? W2 : W3;
  bf16* T = (z == 0) ? T0 : (z == 1) ? T1 : (z == 2) ? T2 : T3;
  int o = blockIdx.x * 256 + threadIdx.x;  // grid.x = 1024*1024/256
  int n = o >> 10, k = o & 1023;
  T[o] = (bf16)W[(size_t)k * 1024 + n];
}

// ---------------- QKV projection GEMM ----------------
// C[M=4096, N=1024] = X[M,K=1024] * W[K,N] + bias[N]; bf16 WMMA, f32 acc.
// Block tile 64(M) x 256(N), 8 waves of 32x64 tiles. Double-buffered LDS with
// async global->LDS staging of k-slice (it+1) overlapped with WMMAs on slice it.
// z=0 -> Q scattered to [B,H,S,D]; z=1 -> K same; z=2 -> V scattered to [B,H,D,S].
__global__ __launch_bounds__(256) void qkv_gemm_kernel(
    const bf16* __restrict__ X,
    const bf16* __restrict__ Wqt, const bf16* __restrict__ Wkt, const bf16* __restrict__ Wvt,
    const float* __restrict__ bq, const float* __restrict__ bk, const float* __restrict__ bv,
    bf16* __restrict__ Qb, bf16* __restrict__ Kb, bf16* __restrict__ Vt) {
  constexpr int BM = 64, BN = 256, BK = 32, NK = Fc / BK;
  constexpr int LDT = 40;  // 80B row stride: 16B aligned, conflict-free bank rotation
  __shared__ __attribute__((aligned(16))) bf16 sA[2][BM][LDT];
  __shared__ __attribute__((aligned(16))) bf16 sB[2][BN][LDT];

  int z = blockIdx.z;
  const bf16* Wt = (z == 0) ? Wqt : (z == 1) ? Wkt : Wvt;
  const float* bias = (z == 0) ? bq : (z == 1) ? bk : bv;

  int tid = threadIdx.x, lane = tid & 31, wave = tid >> 5;
  int wm = wave >> 2, wn = wave & 3;  // 2 x 4 wave grid; wave tile 32(M) x 64(N)
  int m0 = blockIdx.y * BM, n0 = blockIdx.x * BN;

  v8f acc[2][4] = {};
  int arow = tid >> 2, akc = (tid & 3) << 3;  // 64 rows * 4 chunks = 256

  auto stage = [&](int buf, int kt) {
    cp_b128(&sA[buf][arow][akc], X + (size_t)(m0 + arow) * Fc + kt + akc);
#pragma unroll
    for (int i = 0; i < 4; ++i) {  // 256 rows * 4 chunks = 1024
      int c = tid + 256 * i;
      int br = c >> 2, bkc = (c & 3) << 3;
      cp_b128(&sB[buf][br][bkc], Wt + (size_t)(n0 + br) * Fc + kt + bkc);
    }
    if (kt + BK < Fc) {  // L2 prefetch one slice further ahead
      __builtin_prefetch(X + (size_t)(m0 + arow) * Fc + kt + BK + akc, 0, 1);
      __builtin_prefetch(Wt + (size_t)(n0 + arow) * Fc + kt + BK + akc, 0, 1);
    }
  };

  stage(0, 0);
  int buf = 0;
  for (int it = 0; it < NK; ++it) {
    wait_async_copies();   // this wave's staged copies complete
    __syncthreads();       // all waves' copies visible
    if (it + 1 < NK) stage(buf ^ 1, (it + 1) * BK);  // overlap with compute

    v16bf af[2], bfr[4];
#pragma unroll
    for (int mi = 0; mi < 2; ++mi)
      af[mi] = load_frag(&sA[buf][wm * 32 + mi * 16][0], LDT, lane);
#pragma unroll
    for (int ni = 0; ni < 4; ++ni)
      bfr[ni] = load_frag(&sB[buf][wn * 64 + ni * 16][0], LDT, lane);
#pragma unroll
    for (int mi = 0; mi < 2; ++mi)
#pragma unroll
      for (int ni = 0; ni < 4; ++ni) acc[mi][ni] = WMMA_BF16(af[mi], bfr[ni], acc[mi][ni]);
    buf ^= 1;
  }

  int col = lane & 15, rbase = (lane >> 4) << 3;
#pragma unroll
  for (int mi = 0; mi < 2; ++mi)
#pragma unroll
    for (int ni = 0; ni < 4; ++ni) {
      int n = n0 + wn * 64 + ni * 16 + col;
      int h = n >> 6, d = n & 63;
      float bi = bias[n];
#pragma unroll
      for (int j = 0; j < 8; ++j) {
        int rowg = m0 + wm * 32 + mi * 16 + rbase + j;
        int b = rowg >> 10, s = rowg & 1023;
        bf16 val = (bf16)(acc[mi][ni][j] + bi);
        if (z == 0)
          Qb[((size_t)((b * Hc + h) * Sc) + s) * Dc + d] = val;
        else if (z == 1)
          Kb[((size_t)((b * Hc + h) * Sc) + s) * Dc + d] = val;
        else
          Vt[((size_t)((b * Hc + h) * Dc) + d) * Sc + s] = val;
      }
    }
}

// ---------------- fused synthesizer attention ----------------
// One wave per 16-row q-tile; kv streamed in 32-column steps (flash-style).
__global__ __launch_bounds__(128) void attn_kernel(
    const bf16* __restrict__ Qb, const bf16* __restrict__ Kb,
    const bf16* __restrict__ Vt, const float* __restrict__ R,
    bf16* __restrict__ Yb) {
  __shared__ __attribute__((aligned(16))) bf16 sP[4][16][32];  // per-wave P transpose buffer
  int tid = threadIdx.x, lane = tid & 31, wave = tid >> 5;
  int h = blockIdx.y, b = blockIdx.z;
  int q0 = (blockIdx.x * 4 + wave) * 16;

  const bf16* Q = Qb + (size_t)((b * Hc + h) * Sc) * Dc;  // [S][64]
  const bf16* K = Kb + (size_t)((b * Hc + h) * Sc) * Dc;  // [S][64]
  const bf16* V = Vt + (size_t)((b * Hc + h) * Dc) * Sc;  // [64][S]
  const float* Rh = R + (size_t)h * Sc * Sc;              // [q][k]

  int col = lane & 15;
  int rbase = (lane >> 4) << 3;

  v16bf qf0 = load_frag(Q + (size_t)q0 * Dc, Dc, lane);       // d 0..31
  v16bf qf1 = load_frag(Q + (size_t)q0 * Dc + 32, Dc, lane);  // d 32..63

  v8f o[4] = {};
  float mrow[8], lrow[8];
#pragma unroll
  for (int j = 0; j < 8; ++j) { mrow[j] = -1e30f; lrow[j] = 0.f; }
  const float scale = 0.125f;  // 1/sqrt(64)

  for (int kv0 = 0; kv0 < Sc; kv0 += 32) {
    if (kv0 + 32 < Sc) {  // prefetch next kv tile (K rows, V rows, R lines)
      __builtin_prefetch(K + (size_t)(kv0 + 32 + (lane & 31)) * Dc, 0, 1);
      __builtin_prefetch(V + (size_t)(lane & 15) * Sc + kv0 + 32, 0, 1);
      __builtin_prefetch(Rh + (size_t)(q0 + (lane & 15)) * Sc + kv0 + 32, 0, 1);
    }
    v16bf k00 = load_frag(K + (size_t)kv0 * Dc, Dc, lane);
    v16bf k01 = load_frag(K + (size_t)kv0 * Dc + 32, Dc, lane);
    v16bf k10 = load_frag(K + (size_t)(kv0 + 16) * Dc, Dc, lane);
    v16bf k11 = load_frag(K + (size_t)(kv0 + 16) * Dc + 32, Dc, lane);
    v8f sc0 = {}, sc1 = {};
    sc0 = WMMA_BF16(qf0, k00, sc0);
    sc0 = WMMA_BF16(qf1, k01, sc0);
    sc1 = WMMA_BF16(qf0, k10, sc1);
    sc1 = WMMA_BF16(qf1, k11, sc1);

#pragma unroll
    for (int j = 0; j < 8; ++j) {
      const float* rp = Rh + (size_t)(q0 + rbase + j) * Sc + kv0 + col;
      float s0 = sc0[j] * scale + rp[0];
      float s1 = sc1[j] * scale + rp[16];
      float mx = fmaxf(s0, s1);
#pragma unroll
      for (int off = 1; off < 16; off <<= 1) mx = fmaxf(mx, __shfl_xor(mx, off, 16));
      float mnew = fmaxf(mrow[j], mx);
      float alpha = __expf(mrow[j] - mnew);
      float p0 = __expf(s0 - mnew);
      float p1 = __expf(s1 - mnew);
      float rs = p0 + p1;
#pragma unroll
      for (int off = 1; off < 16; off <<= 1) rs += __shfl_xor(rs, off, 16);
      lrow[j] = lrow[j] * alpha + rs;
      mrow[j] = mnew;
#pragma unroll
      for (int t = 0; t < 4; ++t) o[t][j] *= alpha;
      sP[wave][rbase + j][col] = (bf16)p0;
      sP[wave][rbase + j][16 + col] = (bf16)p1;
    }

    // P (16x32) as A-fragment via per-wave LDS transpose; V^T rows are contiguous kv.
    v16bf pf = load_frag(&sP[wave][0][0], 32, lane);
#pragma unroll
    for (int t = 0; t < 4; ++t) {
      v16bf vf = load_frag(V + (size_t)(t * 16) * Sc + kv0, Sc, lane);
      o[t] = WMMA_BF16(pf, vf, o[t]);
    }
  }

#pragma unroll
  for (int t = 0; t < 4; ++t)
#pragma unroll
    for (int j = 0; j < 8; ++j) {
      float val = o[t][j] / lrow[j];
      Yb[(size_t)(b * Sc + q0 + rbase + j) * (Hc * Dc) + h * Dc + t * 16 + col] = (bf16)val;
    }
}

// ---------------- output projection GEMM ----------------
__global__ __launch_bounds__(256) void out_gemm_kernel(
    const bf16* __restrict__ Yb, const bf16* __restrict__ Wot,
    const float* __restrict__ bo, float* __restrict__ out) {
  constexpr int BM = 64, BN = 256, BK = 32, NK = Fc / BK;
  constexpr int LDT = 40;
  __shared__ __attribute__((aligned(16))) bf16 sA[2][BM][LDT];
  __shared__ __attribute__((aligned(16))) bf16 sB[2][BN][LDT];

  int tid = threadIdx.x, lane = tid & 31, wave = tid >> 5;
  int wm = wave >> 2, wn = wave & 3;
  int m0 = blockIdx.y * BM, n0 = blockIdx.x * BN;

  v8f acc[2][4] = {};
  int arow = tid >> 2, akc = (tid & 3) << 3;

  auto stage = [&](int buf, int kt) {
    cp_b128(&sA[buf][arow][akc], Yb + (size_t)(m0 + arow) * Fc + kt + akc);
#pragma unroll
    for (int i = 0; i < 4; ++i) {
      int c = tid + 256 * i;
      int br = c >> 2, bkc = (c & 3) << 3;
      cp_b128(&sB[buf][br][bkc], Wot + (size_t)(n0 + br) * Fc + kt + bkc);
    }
    if (kt + BK < Fc) {
      __builtin_prefetch(Yb + (size_t)(m0 + arow) * Fc + kt + BK + akc, 0, 1);
      __builtin_prefetch(Wot + (size_t)(n0 + arow) * Fc + kt + BK + akc, 0, 1);
    }
  };

  stage(0, 0);
  int buf = 0;
  for (int it = 0; it < NK; ++it) {
    wait_async_copies();
    __syncthreads();
    if (it + 1 < NK) stage(buf ^ 1, (it + 1) * BK);

    v16bf af[2], bfr[4];
#pragma unroll
    for (int mi = 0; mi < 2; ++mi)
      af[mi] = load_frag(&sA[buf][wm * 32 + mi * 16][0], LDT, lane);
#pragma unroll
    for (int ni = 0; ni < 4; ++ni)
      bfr[ni] = load_frag(&sB[buf][wn * 64 + ni * 16][0], LDT, lane);
#pragma unroll
    for (int mi = 0; mi < 2; ++mi)
#pragma unroll
      for (int ni = 0; ni < 4; ++ni) acc[mi][ni] = WMMA_BF16(af[mi], bfr[ni], acc[mi][ni]);
    buf ^= 1;
  }

  int col = lane & 15, rbase = (lane >> 4) << 3;
#pragma unroll
  for (int mi = 0; mi < 2; ++mi)
#pragma unroll
    for (int ni = 0; ni < 4; ++ni) {
      int n = n0 + wn * 64 + ni * 16 + col;
      float bi = bo[n];
#pragma unroll
      for (int j = 0; j < 8; ++j) {
        int rowg = m0 + wm * 32 + mi * 16 + rbase + j;
        out[(size_t)rowg * Fc + n] = acc[mi][ni][j] + bi;
      }
    }
}

// ---------------- launcher ----------------
extern "C" void kernel_launch(void* const* d_in, const int* in_sizes, int n_in,
                              void* d_out, int out_size, void* d_ws, size_t ws_size,
                              hipStream_t stream) {
  const float* X  = (const float*)d_in[0];
  const float* Wq = (const float*)d_in[1];
  const float* bq = (const float*)d_in[2];
  const float* Wk = (const float*)d_in[3];
  const float* bk = (const float*)d_in[4];
  const float* Wv = (const float*)d_in[5];
  const float* bv = (const float*)d_in[6];
  const float* R  = (const float*)d_in[7];
  const float* Wo = (const float*)d_in[8];
  const float* bo = (const float*)d_in[9];
  float* out = (float*)d_out;

  char* ws = (char*)d_ws;
  const size_t MB = (size_t)1 << 20;
  bf16* Xb  = (bf16*)(ws + 0 * MB);   // [4096][1024] bf16 inputs
  bf16* Wqt = (bf16*)(ws + 8 * MB);   // [N=1024][K=1024] transposed bf16 weights
  bf16* Wkt = (bf16*)(ws + 10 * MB);
  bf16* Wvt = (bf16*)(ws + 12 * MB);
  bf16* Wot = (bf16*)(ws + 14 * MB);
  bf16* Qb  = (bf16*)(ws + 16 * MB);  // [B,H,S,D]
  bf16* Kb  = (bf16*)(ws + 24 * MB);  // [B,H,S,D]
  bf16* Vt  = (bf16*)(ws + 32 * MB);  // [B,H,D,S]
  bf16* Yb  = (bf16*)(ws + 40 * MB);  // [4096][1024]

  cvt_bf16_kernel<<<dim3((BSc * Fc / 4) / 256), 256, 0, stream>>>(X, Xb, BSc * Fc / 4);
  transpose_cvt_kernel<<<dim3(1024 * 1024 / 256, 1, 4), 256, 0, stream>>>(
      Wq, Wk, Wv, Wo, Wqt, Wkt, Wvt, Wot);
  qkv_gemm_kernel<<<dim3(Fc / 256, BSc / 64, 3), 256, 0, stream>>>(
      Xb, Wqt, Wkt, Wvt, bq, bk, bv, Qb, Kb, Vt);
  attn_kernel<<<dim3(Sc / 64, Hc, Bc), 128, 0, stream>>>(Qb, Kb, Vt, R, Yb);
  out_gemm_kernel<<<dim3(Fc / 256, BSc / 64), 256, 0, stream>>>(Yb, Wot, bo, out);
}
